// MixtureCDFFLowCoupling_62285615727068
// MI455X (gfx1250) — matrix operand order repaired and checked by reading, not compiled, not run on hardware
//
#include <hip/hip_runtime.h>
#include <hip/hip_bf16.h>
#include <math.h>

// ---------------------------------------------------------------------------
// MixtureCDFFlowCoupling, fused for MI455X (gfx1250, wave32, WMMA bf16)
//   SHAPE=64, K=8, HID=512, OUT=1160 (pad->1168), N = 32*2048 = 65536 tokens
// ---------------------------------------------------------------------------

typedef __attribute__((ext_vector_type(16))) __bf16 v16bf;
typedef __attribute__((ext_vector_type(8)))  __bf16 v8bf;
typedef __attribute__((ext_vector_type(8)))  float  v8f;

#define HID      512
#define OUTP     1168          // padded OUT (1160 -> 73*16)
#define OUTR     1160
#define NTILE3   73            // OUTP/16
#define MTILE    16            // tokens per block
#define NTOK     65536
#define N64      (NTOK * 64)

__device__ __forceinline__ v8f vsplat8(float f) {
    return (v8f){f, f, f, f, f, f, f, f};
}

__device__ __forceinline__ v16bf cat16(v8bf lo, v8bf hi) {
    return __builtin_shufflevector(lo, hi, 0,1,2,3,4,5,6,7,8,9,10,11,12,13,14,15);
}

// A-operand for 16x16x32 bf16 WMMA: row r = lane&15, two 8-element K chunks
// (lanes 0-15: K 0..7 & 16..23, lanes 16-31: K 8..15 & 24..31)
__device__ __forceinline__ v16bf load_a(const __bf16* p) {
    v8bf lo = *(const v8bf*)(p);
    v8bf hi = *(const v8bf*)(p + 16);
    return cat16(lo, hi);
}

__device__ __forceinline__ v8f wmma_bf16(v16bf a, v16bf b, v8f c) {
    return __builtin_amdgcn_wmma_f32_16x16x32_bf16(
        /*neg_a=*/false, a, /*neg_b=*/false, b,
        /*c_mod=*/(short)0, c, /*reuse_a=*/false, /*reuse_b=*/false);
}

// ------------------------- weight conversion kernels -----------------------
// W1 [64][512] -> W1t [512][32] bf16 (only masked-active K rows 0..31)
__global__ void k_w1t(const float* __restrict__ W1, __bf16* __restrict__ W1t) {
    int i = blockIdx.x * 256 + threadIdx.x;          // 512*32
    if (i < 512 * 32) {
        int n = i >> 5, k = i & 31;
        W1t[i] = (__bf16)W1[k * 512 + n];
    }
}
// W2 [512][512] -> W2t [512][512] bf16
__global__ void k_w2t(const float* __restrict__ W2, __bf16* __restrict__ W2t) {
    int i = blockIdx.x * 256 + threadIdx.x;          // 512*512
    int n = i >> 9, k = i & 511;
    W2t[i] = (__bf16)W2[k * 512 + n];
}
// W3 [512][1160] -> W3t [1168][512] bf16, zero-padded rows
__global__ void k_w3t(const float* __restrict__ W3, __bf16* __restrict__ W3t) {
    int i = blockIdx.x * 256 + threadIdx.x;          // 1168*512
    if (i < OUTP * 512) {
        int n = i >> 9, k = i & 511;
        W3t[i] = (n < OUTR) ? (__bf16)W3[k * OUTR + n] : (__bf16)0.0f;
    }
}

// ------------------------------ fused main kernel --------------------------
__global__ __launch_bounds__(256)
void coupling_fused(const float* __restrict__ x,
                    const float* __restrict__ b1,
                    const float* __restrict__ b2,
                    const float* __restrict__ b3,
                    const __bf16* __restrict__ W1t,
                    const __bf16* __restrict__ W2t,
                    const __bf16* __restrict__ W3t,
                    float* __restrict__ out) {
    __shared__ float  s_x[MTILE * 64];        //  4 KB  raw input tile (f32)
    __shared__ __bf16 s_xa[MTILE * 32];       //  1 KB  masked bf16 A (K=32)
    __shared__ __bf16 s_h1[MTILE * HID];      // 16 KB
    __shared__ __bf16 s_h2[MTILE * HID];      // 16 KB
    __shared__ float  s_params[MTILE * OUTP]; // 73 KB
    __shared__ float  s_wts[MTILE * 8];       // softmax weights

    const int tid  = threadIdx.x;
    const int wv   = tid >> 5;                // wave id 0..7
    const int lane = tid & 31;
    const int r    = lane & 15;               // row (A) / column-in-tile (B,C)
    const int hi   = lane >> 4;               // lane-half selector
    const long tok0 = (long)blockIdx.x * MTILE;

    // ---- stage input tile; build masked bf16 A matrix --------------------
    #pragma unroll
    for (int rep = 0; rep < 4; ++rep) {
        int idx = rep * 256 + tid;            // 0..1023
        int tk = idx >> 6, d = idx & 63;
        float v = x[(tok0 + tk) * 64 + d];
        s_x[idx] = v;
        if (d < 32) s_xa[tk * 32 + d] = (__bf16)v;
    }
    __syncthreads();

    // ---- layer 1: [16x32] @ [32x512] (mask folds K 64->32) ---------------
    {
        const __bf16* ap = s_xa + r * 32 + hi * 8;
        v16bf a = load_a(ap);
        #pragma unroll
        for (int t = 0; t < 4; ++t) {
            int col = (wv * 4 + t) * 16 + r;
            v16bf b = load_a(W1t + col * 32 + hi * 8);
            v8f acc = vsplat8(b1[col]);
            acc = wmma_bf16(a, b, acc);
            #pragma unroll
            for (int i = 0; i < 8; ++i) {
                float v = acc[i];
                s_h1[(i + 8 * hi) * HID + col] = (__bf16)(v > 0.f ? v : 0.f);
            }
        }
    }
    __syncthreads();

    // ---- layer 2: [16x512] @ [512x512], 16 k-steps, 4 tiles/wave ----------
    {
        v8f acc[4];
        #pragma unroll
        for (int t = 0; t < 4; ++t)
            acc[t] = vsplat8(b2[(wv * 4 + t) * 16 + r]);
        for (int ks = 0; ks < 16; ++ks) {
            const __bf16* ap = s_h1 + r * HID + ks * 32 + hi * 8;
            v16bf a = load_a(ap);
            #pragma unroll
            for (int t = 0; t < 4; ++t) {
                int col = (wv * 4 + t) * 16 + r;
                v16bf b = load_a(W2t + (size_t)col * HID + ks * 32 + hi * 8);
                acc[t] = wmma_bf16(a, b, acc[t]);
            }
        }
        #pragma unroll
        for (int t = 0; t < 4; ++t) {
            int col = (wv * 4 + t) * 16 + r;
            #pragma unroll
            for (int i = 0; i < 8; ++i) {
                float v = acc[t][i];
                s_h2[(i + 8 * hi) * HID + col] = (__bf16)(v > 0.f ? v : 0.f);
            }
        }
    }
    __syncthreads();

    // ---- layer 3: [16x512] @ [512x1168], 73 tiles round-robin -------------
    for (int jt = wv * 4; jt < NTILE3; jt += 32) {   // wave-uniform
        v8f acc[4];
        #pragma unroll
        for (int t = 0; t < 4; ++t) {
            int ct = jt + t, col = ct * 16 + r;
            acc[t] = vsplat8((ct < NTILE3 && col < OUTR) ? b3[col] : 0.f);
        }
        for (int ks = 0; ks < 16; ++ks) {
            const __bf16* ap = s_h2 + r * HID + ks * 32 + hi * 8;
            v16bf a = load_a(ap);
            #pragma unroll
            for (int t = 0; t < 4; ++t) {
                int ct = jt + t;
                if (ct < NTILE3) {                   // wave-uniform guard
                    v16bf b = load_a(W3t + (size_t)(ct * 16 + r) * HID
                                                  + ks * 32 + hi * 8);
                    acc[t] = wmma_bf16(a, b, acc[t]);
                }
            }
        }
        #pragma unroll
        for (int t = 0; t < 4; ++t) {
            int ct = jt + t;
            if (ct < NTILE3) {
                int col = ct * 16 + r;
                #pragma unroll
                for (int i = 0; i < 8; ++i)
                    s_params[(i + 8 * hi) * OUTP + col] = acc[t][i];
            }
        }
    }
    __syncthreads();

    // ---- per-token softmax over 8 mixture logits --------------------------
    if (tid < MTILE) {
        const float* lg = s_params + tid * OUTP + 128;
        float mx = lg[0];
        #pragma unroll
        for (int k = 1; k < 8; ++k) mx = fmaxf(mx, lg[k]);
        float e[8], sum = 0.f;
        #pragma unroll
        for (int k = 0; k < 8; ++k) { e[k] = __expf(lg[k] - mx); sum += e[k]; }
        float inv = 1.f / sum;
        #pragma unroll
        for (int k = 0; k < 8; ++k) s_wts[tid * 8 + k] = e[k] * inv;
    }
    __syncthreads();

    // ---- mixture-CDF flow tail (f32) --------------------------------------
    #pragma unroll
    for (int rep = 0; rep < 4; ++rep) {
        int idx = rep * 256 + tid;            // 0..1023  -> (token, dim)
        int tk = idx >> 6, d = idx & 63;
        float xv = s_x[tk * 64 + d];
        float xo, ld;
        if (d < 32) {                         // masked: pass-through
            xo = xv;
            ld = 0.f;
        } else {
            const float* pp = s_params + tk * OUTP;
            float logscale = pp[d];
            float bias     = pp[64 + d];
            float z = 0.f, pmix = 0.f;
            #pragma unroll
            for (int k = 0; k < 8; ++k) {
                float mu   = pp[136 + k * 64 + d];
                float lstd = pp[648 + k * 64 + d];
                float istd = __expf(-lstd);
                float u    = (xv - mu) * istd;
                float wk   = s_wts[tk * 8 + k];
                z    += wk * 0.5f * (1.f + erff(u * 0.70710678118654752f));
                pmix += wk * __expf(-0.5f * u * u) * istd * 0.39894228040143268f;
            }
            float lz  = __logf(z);
            float l1z = __logf(1.f - z);
            xo = (lz - l1z) * __expf(logscale) + bias;
            ld = __logf(pmix) - lz - l1z + logscale;
        }
        long o = (tok0 + tk) * 64 + d;
        out[o]       = xo;                    // x_out
        out[N64 + o] = ld;                    // log_det
    }
}

// ---------------------------------------------------------------------------
extern "C" void kernel_launch(void* const* d_in, const int* in_sizes, int n_in,
                              void* d_out, int out_size, void* d_ws, size_t ws_size,
                              hipStream_t stream) {
    const float* x  = (const float*)d_in[0];
    const float* W1 = (const float*)d_in[1];
    const float* b1 = (const float*)d_in[2];
    const float* W2 = (const float*)d_in[3];
    const float* b2 = (const float*)d_in[4];
    const float* W3 = (const float*)d_in[5];
    const float* b3 = (const float*)d_in[6];
    float* out = (float*)d_out;

    // workspace layout (bf16 transposed weights)
    char* ws = (char*)d_ws;
    __bf16* W1t = (__bf16*)(ws);                               //  32 KB
    __bf16* W2t = (__bf16*)(ws + (64 * 1024));                 // 512 KB
    __bf16* W3t = (__bf16*)(ws + (64 * 1024) + (512 * 1024));  // ~1.17 MB

    k_w1t<<<(512 * 32 + 255) / 256, 256, 0, stream>>>(W1, W1t);
    k_w2t<<<(512 * 512) / 256,      256, 0, stream>>>(W2, W2t);
    k_w3t<<<(OUTP * 512 + 255) / 256, 256, 0, stream>>>(W3, W3t);

    coupling_fused<<<NTOK / MTILE, 256, 0, stream>>>(x, b1, b2, b3,
                                                     W1t, W2t, W3t, out);
}
